// MultiHeadAttention_45543833206817
// MI455X (gfx1250) — matrix operand order repaired and checked by reading, not compiled, not run on hardware
//
#include <hip/hip_runtime.h>

#define BATCH 4
#define SEQ   2048
#define EMB   1024
#define HEADS 16
#define HDIM  64
#define QK_SCALE 0.125f   // 64^-0.5

typedef __attribute__((ext_vector_type(16))) __bf16 v16bf;
typedef __attribute__((ext_vector_type(8)))  float  v8f;

union BF16x16 { v16bf v; uint4 q[2]; };

// exact pointee type of the async-LDS builtin params (per compiler diagnostic)
typedef int v4i_t __attribute__((vector_size(16)));
typedef __attribute__((address_space(1))) v4i_t GLB_V4I;  // global
typedef __attribute__((address_space(3))) v4i_t LDS_V4I;  // LDS

static __device__ __forceinline__ unsigned short f2bf(float f) {
  union { float f; unsigned int u; } v;
  v.f = f;
  unsigned int r = v.u + 0x7FFFu + ((v.u >> 16) & 1u);  // round-to-nearest-even
  return (unsigned short)(r >> 16);
}

// ---- async global->LDS staging (CDNA5 GLOBAL_LOAD_ASYNC_TO_LDS_B128),
// ---- with a synchronous fallback if the builtin is absent on this toolchain.
#if defined(__has_builtin)
#if __has_builtin(__builtin_amdgcn_global_load_async_to_lds_b128)
#define HAVE_ASYNC_LDS 1
#endif
#if __has_builtin(__builtin_amdgcn_s_wait_asynccnt)
#define HAVE_ASYNC_WAIT 1
#endif
#endif

static __device__ __forceinline__ void g2lds16(const unsigned short* g,
                                               unsigned short* l) {
#if defined(HAVE_ASYNC_LDS)
  __builtin_amdgcn_global_load_async_to_lds_b128(
      (GLB_V4I*)(unsigned long long)g,
      (LDS_V4I*)(unsigned)(unsigned long long)l, 0, 0);
#else
  *(uint4*)l = *(const uint4*)g;
#endif
}

#if defined(HAVE_ASYNC_LDS) && defined(HAVE_ASYNC_WAIT)
#define ASYNC_WAIT(n) __builtin_amdgcn_s_wait_asynccnt(n)
#else
#define ASYNC_WAIT(n)
#endif

// ---------------------------------------------------------------------------
// 1) LayerNorm -> bf16
// ---------------------------------------------------------------------------
__global__ void ln_bf16_kernel(const float* __restrict__ x,
                               const float* __restrict__ w,
                               const float* __restrict__ b,
                               unsigned short* __restrict__ hb) {
  __shared__ float red[16];
  int row = blockIdx.x;
  const float* xr = x + (size_t)row * EMB;
  float vals[4];
  float s = 0.f, s2 = 0.f;
#pragma unroll
  for (int i = 0; i < 4; ++i) {
    float v = xr[threadIdx.x + i * 256];
    vals[i] = v; s += v; s2 += v * v;
  }
#pragma unroll
  for (int off = 16; off >= 1; off >>= 1) {
    s  += __shfl_xor(s,  off, 32);
    s2 += __shfl_xor(s2, off, 32);
  }
  int wid = threadIdx.x >> 5;
  if ((threadIdx.x & 31) == 0) { red[wid] = s; red[8 + wid] = s2; }
  __syncthreads();
  s = 0.f; s2 = 0.f;
#pragma unroll
  for (int i = 0; i < 8; ++i) { s += red[i]; s2 += red[8 + i]; }
  float mu   = s * (1.0f / EMB);
  float var  = s2 * (1.0f / EMB) - mu * mu;
  float rstd = rsqrtf(var + 1e-5f);
#pragma unroll
  for (int i = 0; i < 4; ++i) {
    int c = threadIdx.x + i * 256;
    float h = (vals[i] - mu) * rstd * w[c] + b[c];
    hb[(size_t)row * EMB + c] = f2bf(h);
  }
}

// ---------------------------------------------------------------------------
// 2) Transpose fp32 [K][C] -> bf16 [C][K]
// ---------------------------------------------------------------------------
__global__ void transpose_bf16_kernel(const float* __restrict__ src,
                                      unsigned short* __restrict__ dst,
                                      int K, int C) {
  size_t total = (size_t)K * C;
  for (size_t i = (size_t)blockIdx.x * blockDim.x + threadIdx.x; i < total;
       i += (size_t)gridDim.x * blockDim.x) {
    size_t c = i / K, k = i % K;
    dst[i] = f2bf(src[k * (size_t)C + c]);
  }
}

// ---------------------------------------------------------------------------
// 3) QKV GEMM, register-blocked 64x32 per wave (4x2 WMMA tiles).
//    Scatter into Q,K ([b][h][n][d]) and V transposed ([b][h][d][n]).
// ---------------------------------------------------------------------------
__global__ void qkv_gemm_kernel(const unsigned short* __restrict__ hb,
                                const unsigned short* __restrict__ wT,
                                const float* __restrict__ bqkv,
                                unsigned short* __restrict__ Qb,
                                unsigned short* __restrict__ Kb,
                                unsigned short* __restrict__ Vt) {
  const int NT2 = (3 * EMB) / 32;  // 96 column blocks of 32
  int wave = blockIdx.x * (blockDim.x >> 5) + (threadIdx.x >> 5);
  int lane = threadIdx.x & 31;
  int g = lane >> 4, m = lane & 15;
  int tm = (wave / NT2) * 64;
  int tn = (wave % NT2) * 32;
  const unsigned short* ar[4];
  const unsigned short* br[2];
#pragma unroll
  for (int i = 0; i < 4; ++i) ar[i] = hb + (size_t)(tm + 16 * i + m) * EMB;
#pragma unroll
  for (int jj = 0; jj < 2; ++jj) br[jj] = wT + (size_t)(tn + 16 * jj + m) * EMB;

  v8f zero = {};
  v8f acc[4][2];
#pragma unroll
  for (int i = 0; i < 4; ++i)
#pragma unroll
    for (int jj = 0; jj < 2; ++jj) acc[i][jj] = zero;

  for (int k0 = 0; k0 < EMB; k0 += 32) {
    BF16x16 a[4], b[2];
#pragma unroll
    for (int i = 0; i < 4; ++i) {
      a[i].q[0] = *(const uint4*)(ar[i] + k0 + 8 * g);
      a[i].q[1] = *(const uint4*)(ar[i] + k0 + 16 + 8 * g);
    }
#pragma unroll
    for (int jj = 0; jj < 2; ++jj) {
      b[jj].q[0] = *(const uint4*)(br[jj] + k0 + 8 * g);
      b[jj].q[1] = *(const uint4*)(br[jj] + k0 + 16 + 8 * g);
    }
#pragma unroll
    for (int i = 0; i < 4; ++i)
#pragma unroll
      for (int jj = 0; jj < 2; ++jj)
        acc[i][jj] = __builtin_amdgcn_wmma_f32_16x16x32_bf16(
            false, a[i].v, false, b[jj].v, (short)0, acc[i][jj], false, false);
  }

#pragma unroll
  for (int jj = 0; jj < 2; ++jj) {
    int col   = tn + 16 * jj + m;       // column in [0, 3E)
    int h     = col / (3 * HDIM);
    int rem   = col % (3 * HDIM);
    int d     = rem / 3;
    int which = rem % 3;                // 0=q 1=k 2=v (interleaved fastest)
    float bias = bqkv[col];
#pragma unroll
    for (int i = 0; i < 4; ++i) {
#pragma unroll
      for (int r = 0; r < 8; ++r) {
        int row = tm + 16 * i + r + 8 * g;
        int bb = row / SEQ, n = row % SEQ;
        unsigned short val = f2bf(acc[i][jj][r] + bias);
        size_t bhn = (size_t)(bb * HEADS + h);
        if (which == 0)      Qb[(bhn * SEQ + n) * HDIM + d] = val;
        else if (which == 1) Kb[(bhn * SEQ + n) * HDIM + d] = val;
        else                 Vt[(bhn * HDIM + d) * SEQ + n] = val;
      }
    }
  }
}

// ---------------------------------------------------------------------------
// 4) Flash attention. 8 waves/block share one (b,h); K/V blocks are staged in
//    double-buffered LDS via async global->LDS DMA. One 16-query tile/wave.
// ---------------------------------------------------------------------------
#define KT 32
__global__ void attn_kernel(const unsigned short* __restrict__ Qb,
                            const unsigned short* __restrict__ Kb,
                            const unsigned short* __restrict__ Vt,
                            unsigned short* __restrict__ attO) {
  __shared__ unsigned short Kbuf[2][KT][HDIM];   // 8 KB  [key][d]
  __shared__ unsigned short Vbuf[2][HDIM][KT];   // 8 KB  [d][key]  (V^T)
  __shared__ unsigned short ptile[8][16][KT];    // 8 KB  P per wave
  int wid  = threadIdx.x >> 5;
  int lane = threadIdx.x & 31;
  int g = lane >> 4, m = lane & 15;
  int bh = blockIdx.x >> 4;                  // 16 blocks per (b,h)
  int qt = (blockIdx.x & 15) * 8 + wid;      // qtile in [0,128)
  const unsigned short* qbase = Qb + ((size_t)bh * SEQ + (size_t)qt * 16) * HDIM;
  const unsigned short* kbase = Kb + (size_t)bh * SEQ * HDIM;
  const unsigned short* vbase = Vt + (size_t)bh * HDIM * SEQ;

  // cooperative staging slices (each wave moves 512B of K + 512B of V)
  int kr = wid * 4 + (lane >> 3), kc = (lane & 7) * 8;  // K: 32 rows x 64
  int vr = wid * 8 + (lane >> 2), vc = (lane & 3) * 8;  // V^T: 64 rows x 32

  // Q A-fragments: row = m, two K-chunks of 32 over d=64
  BF16x16 qa[2];
  const unsigned short* qrow = qbase + (size_t)m * HDIM;
#pragma unroll
  for (int kk = 0; kk < 2; ++kk) {
    qa[kk].q[0] = *(const uint4*)(qrow + kk * 32 + 8 * g);
    qa[kk].q[1] = *(const uint4*)(qrow + kk * 32 + 16 + 8 * g);
  }

  v8f zero = {};
  v8f o[4];
  float mrow[8], lrow[8];
#pragma unroll
  for (int r = 0; r < 8; ++r) { mrow[r] = -1e30f; lrow[r] = 0.f; }
#pragma unroll
  for (int t = 0; t < 4; ++t) o[t] = zero;

  // prologue: stage first K/V block into buffer 0
  g2lds16(kbase + (size_t)kr * HDIM + kc, &Kbuf[0][kr][kc]);
  g2lds16(vbase + (size_t)vr * SEQ + vc, &Vbuf[0][vr][vc]);

  int p = 0;
  for (int j = 0; j < SEQ; j += KT) {
    if (j + KT < SEQ) {  // issue next block, then wait for current (<=2 left)
      g2lds16(kbase + (size_t)(j + KT + kr) * HDIM + kc, &Kbuf[p ^ 1][kr][kc]);
      g2lds16(vbase + (size_t)vr * SEQ + (j + KT) + vc, &Vbuf[p ^ 1][vr][vc]);
      ASYNC_WAIT(2);
    } else {
      ASYNC_WAIT(0);
    }
    __syncthreads();  // current K/V block visible to all waves

    // ---- scores: S = Q x K^T for keys [j, j+32), K from LDS
    v8f s0 = zero, s1 = zero;
    const unsigned short* krow0 = &Kbuf[p][m][0];
    const unsigned short* krow1 = &Kbuf[p][16 + m][0];
#pragma unroll
    for (int kk = 0; kk < 2; ++kk) {
      BF16x16 b0, b1;
      b0.q[0] = *(const uint4*)(krow0 + kk * 32 + 8 * g);
      b0.q[1] = *(const uint4*)(krow0 + kk * 32 + 16 + 8 * g);
      b1.q[0] = *(const uint4*)(krow1 + kk * 32 + 8 * g);
      b1.q[1] = *(const uint4*)(krow1 + kk * 32 + 16 + 8 * g);
      s0 = __builtin_amdgcn_wmma_f32_16x16x32_bf16(false, qa[kk].v, false, b0.v,
                                                   (short)0, s0, false, false);
      s1 = __builtin_amdgcn_wmma_f32_16x16x32_bf16(false, qa[kk].v, false, b1.v,
                                                   (short)0, s1, false, false);
    }

    // ---- online softmax (row = r + 8g lives in a 16-lane half)
    float fac[8];
#pragma unroll
    for (int r = 0; r < 8; ++r) {
      float sv0 = s0[r] * QK_SCALE, sv1 = s1[r] * QK_SCALE;
      float v = fmaxf(sv0, sv1);
      v = fmaxf(v, __shfl_xor(v, 1, 32));
      v = fmaxf(v, __shfl_xor(v, 2, 32));
      v = fmaxf(v, __shfl_xor(v, 4, 32));
      v = fmaxf(v, __shfl_xor(v, 8, 32));
      float mnew = fmaxf(mrow[r], v);
      float f  = __expf(mrow[r] - mnew);
      float p0 = __expf(sv0 - mnew);
      float p1 = __expf(sv1 - mnew);
      float rs = p0 + p1;
      rs += __shfl_xor(rs, 1, 32);
      rs += __shfl_xor(rs, 2, 32);
      rs += __shfl_xor(rs, 4, 32);
      rs += __shfl_xor(rs, 8, 32);
      lrow[r] = lrow[r] * f + rs;
      mrow[r] = mnew;
      fac[r]  = f;
      ptile[wid][r + 8 * g][m]      = f2bf(p0);
      ptile[wid][r + 8 * g][16 + m] = f2bf(p1);
    }
#pragma unroll
    for (int t = 0; t < 4; ++t)
#pragma unroll
      for (int r = 0; r < 8; ++r) o[t][r] *= fac[r];
    __syncthreads();

    // ---- reload P in A-fragment layout; o += P x V (V from LDS)
    BF16x16 pa;
    pa.q[0] = *(const uint4*)(&ptile[wid][m][8 * g]);
    pa.q[1] = *(const uint4*)(&ptile[wid][m][16 + 8 * g]);
#pragma unroll
    for (int t = 0; t < 4; ++t) {
      BF16x16 vb;
      const unsigned short* vrow = &Vbuf[p][t * 16 + m][8 * g];
      vb.q[0] = *(const uint4*)(vrow);
      vb.q[1] = *(const uint4*)(vrow + 16);
      o[t] = __builtin_amdgcn_wmma_f32_16x16x32_bf16(false, pa.v, false, vb.v,
                                                     (short)0, o[t], false, false);
    }
    __syncthreads();  // buffer p fully consumed; safe to overwrite next iter
    p ^= 1;
  }

  int b = bh / HEADS, h = bh % HEADS;
#pragma unroll
  for (int t = 0; t < 4; ++t)
#pragma unroll
    for (int r = 0; r < 8; ++r) {
      int q = qt * 16 + r + 8 * g;
      int dcol = h * HDIM + t * 16 + m;
      attO[((size_t)b * SEQ + q) * EMB + dcol] = f2bf(o[t][r] / lrow[r]);
    }
}

// ---------------------------------------------------------------------------
// 5) Output projection + bias + residual, register-blocked 64x32 per wave.
// ---------------------------------------------------------------------------
__global__ void proj_kernel(const unsigned short* __restrict__ aO,
                            const unsigned short* __restrict__ wT,
                            const float* __restrict__ bias,
                            const float* __restrict__ resid,
                            float* __restrict__ out) {
  const int NT2 = EMB / 32;  // 32 column blocks
  int wave = blockIdx.x * (blockDim.x >> 5) + (threadIdx.x >> 5);
  int lane = threadIdx.x & 31;
  int g = lane >> 4, m = lane & 15;
  int tm = (wave / NT2) * 64;
  int tn = (wave % NT2) * 32;
  const unsigned short* ar[4];
  const unsigned short* br[2];
#pragma unroll
  for (int i = 0; i < 4; ++i) ar[i] = aO + (size_t)(tm + 16 * i + m) * EMB;
#pragma unroll
  for (int jj = 0; jj < 2; ++jj) br[jj] = wT + (size_t)(tn + 16 * jj + m) * EMB;

  v8f zero = {};
  v8f acc[4][2];
#pragma unroll
  for (int i = 0; i < 4; ++i)
#pragma unroll
    for (int jj = 0; jj < 2; ++jj) acc[i][jj] = zero;

  for (int k0 = 0; k0 < EMB; k0 += 32) {
    BF16x16 a[4], b[2];
#pragma unroll
    for (int i = 0; i < 4; ++i) {
      a[i].q[0] = *(const uint4*)(ar[i] + k0 + 8 * g);
      a[i].q[1] = *(const uint4*)(ar[i] + k0 + 16 + 8 * g);
    }
#pragma unroll
    for (int jj = 0; jj < 2; ++jj) {
      b[jj].q[0] = *(const uint4*)(br[jj] + k0 + 8 * g);
      b[jj].q[1] = *(const uint4*)(br[jj] + k0 + 16 + 8 * g);
    }
#pragma unroll
    for (int i = 0; i < 4; ++i)
#pragma unroll
      for (int jj = 0; jj < 2; ++jj)
        acc[i][jj] = __builtin_amdgcn_wmma_f32_16x16x32_bf16(
            false, a[i].v, false, b[jj].v, (short)0, acc[i][jj], false, false);
  }

#pragma unroll
  for (int jj = 0; jj < 2; ++jj) {
    int col = tn + 16 * jj + m;
    float bval = bias[col];
#pragma unroll
    for (int i = 0; i < 4; ++i) {
#pragma unroll
      for (int r = 0; r < 8; ++r) {
        int row = tm + 16 * i + r + 8 * g;
        size_t idx = (size_t)row * EMB + col;
        out[idx] = acc[i][jj][r] + bval + resid[idx];
      }
    }
  }
}

// ---------------------------------------------------------------------------
// Launch
// ---------------------------------------------------------------------------
extern "C" void kernel_launch(void* const* d_in, const int* in_sizes, int n_in,
                              void* d_out, int out_size, void* d_ws, size_t ws_size,
                              hipStream_t stream) {
  const float* x      = (const float*)d_in[0];
  const float* ln_w   = (const float*)d_in[1];
  const float* ln_b   = (const float*)d_in[2];
  const float* w_qkv  = (const float*)d_in[3];
  const float* b_qkv  = (const float*)d_in[4];
  const float* w_proj = (const float*)d_in[5];
  const float* b_proj = (const float*)d_in[6];
  float* out = (float*)d_out;

  const size_t SZ_ACT    = (size_t)BATCH * SEQ * EMB * 2;  // 16 MB (bf16)
  const size_t OFF_HB    = 0;
  const size_t OFF_WQKVT = OFF_HB    + SZ_ACT;
  const size_t OFF_WPROJ = OFF_WQKVT + (size_t)3 * EMB * EMB * 2;
  const size_t OFF_Q     = OFF_WPROJ + (size_t)EMB * EMB * 2;
  const size_t OFF_K     = OFF_Q     + SZ_ACT;
  const size_t OFF_VT    = OFF_K     + SZ_ACT;
  const size_t OFF_ATTO  = OFF_VT    + SZ_ACT;

  char* ws = (char*)d_ws;
  unsigned short* HB     = (unsigned short*)(ws + OFF_HB);
  unsigned short* WQKVT  = (unsigned short*)(ws + OFF_WQKVT);
  unsigned short* WPROJT = (unsigned short*)(ws + OFF_WPROJ);
  unsigned short* Qb     = (unsigned short*)(ws + OFF_Q);
  unsigned short* Kb     = (unsigned short*)(ws + OFF_K);
  unsigned short* Vt     = (unsigned short*)(ws + OFF_VT);
  unsigned short* ATTO   = (unsigned short*)(ws + OFF_ATTO);

  ln_bf16_kernel<<<BATCH * SEQ, 256, 0, stream>>>(x, ln_w, ln_b, HB);
  transpose_bf16_kernel<<<1024, 256, 0, stream>>>(w_qkv, WQKVT, EMB, 3 * EMB);
  transpose_bf16_kernel<<<512, 256, 0, stream>>>(w_proj, WPROJT, EMB, EMB);

  // QKV GEMM: (8192/64) * (3072/32) = 12288 waves, 8 per block
  qkv_gemm_kernel<<<12288 / 8, 256, 0, stream>>>(HB, WQKVT, b_qkv, Qb, Kb, Vt);

  // Attention: 64 (b,h) x 16 blocks, 8 waves/block (one 16-query tile each)
  attn_kernel<<<64 * 16, 256, 0, stream>>>(Qb, Kb, Vt, ATTO);

  // Projection: (8192/64) * (1024/32) = 4096 waves, 8 per block
  proj_kernel<<<4096 / 8, 256, 0, stream>>>(ATTO, WPROJT, b_proj, x, out);
}
